// GraphNet_4200478015497
// MI455X (gfx1250) — matrix operand order repaired and checked by reading, not compiled, not run on hardware
//
#include <hip/hip_runtime.h>
#include <hip/hip_bf16.h>

// ---------------------------------------------------------------------------
// CGCNN-style graph net for MI455X (gfx1250, wave32, WMMA).
// Heavy GEMMs (1.2M x 169 @ 169 x 128, x3 conv layers + embedding) run on
// v_wmma_f32_16x16x32_bf16 with fp32 accumulation; batchnorm done in two
// passes; small head MLP in scalar fp32 for accuracy.
// ---------------------------------------------------------------------------

#define N_ATOMS 100000
#define M_NBR   12
#define NPAIR   (N_ATOMS * M_NBR)     // 1,200,000
#define N_CRYS  2000
#define ORIG_F  92
#define NBR_F   41
#define ATOM_F  64
#define H_F     128
#define SPUT_F  17
#define BN_EPS  1e-5f

// Conv GEMM geometry: K = 2*64+41 = 169 -> pad to 192 (6 chunks of 32), 128 cols
#define CONV_K    169
#define CONV_KC   6
#define CONV_KP   (CONV_KC * 32)      // 192
#define CONV_LDS_STRIDE (CONV_KP + 8) // pad to avoid LDS bank conflicts (400B rows, 16B aligned)
#define CONV_NT   8                   // 8 col-tiles * 16 = 128
#define ROWS_BLK  128                 // rows per workgroup (8 waves * 16)
#define CONV_BLOCKS (NPAIR / ROWS_BLK) // 9375 exactly

// Embedding GEMM: K = 92 -> pad 96 (3 chunks), 64 cols
#define EMB_K     92
#define EMB_KC    3
#define EMB_KP    (EMB_KC * 32)       // 96
#define EMB_LDS_STRIDE (EMB_KP + 8)   // 104
#define EMB_NT    4
#define EMB_BLOCKS ((N_ATOMS + ROWS_BLK - 1) / ROWS_BLK) // 782

typedef __attribute__((ext_vector_type(16))) __bf16 v16bf;
typedef __attribute__((ext_vector_type(8)))  __bf16 v8bf;
typedef __attribute__((ext_vector_type(8)))  float  v8f;

__device__ __forceinline__ float softplusf(float x) {
    return (x > 20.f) ? x : log1pf(expf(x));
}
__device__ __forceinline__ float sigmoidf(float x) {
    return 1.f / (1.f + expf(-x));
}

// ---------------------------------------------------------------------------
// WMMA inner loop. A staged in LDS (row-major bf16, padded stride). B is
// pre-packed in global memory so each lane's fragment is 16 contiguous bf16:
//   Bp[(((t*KC)+kc)*32 + lane)*16 + e]  ==  W[k = kc*32 + 16*(lane>>4) + e][t*16 + (lane&15)]
// A fragment layout (matches B packing): lane holds row m = lane&15,
// K in {16*kh .. 16*kh+15} contiguous, kh=lane>>4.
// ---------------------------------------------------------------------------
template <int NT, int KC>
__device__ __forceinline__ void wmma_core(const __bf16* ldsA, int ldsStride,
                                          const __bf16* __restrict__ Bp,
                                          v8f acc[NT]) {
    const int lane = threadIdx.x & 31;
    const int wave = threadIdx.x >> 5;
    const int m    = lane & 15;
    const int kh   = lane >> 4;
    const __bf16* arow = ldsA + (wave * 16 + m) * ldsStride;
#pragma unroll
    for (int kc = 0; kc < KC; ++kc) {
        v8bf lo = *(const v8bf*)(arow + kc * 32 + 16 * kh);
        v8bf hi = *(const v8bf*)(arow + kc * 32 + 16 * kh + 8);
        v16bf a;
#pragma unroll
        for (int e = 0; e < 8; ++e) { a[e] = lo[e]; a[e + 8] = hi[e]; }
#pragma unroll
        for (int t = 0; t < NT; ++t) {
            v16bf b = *(const v16bf*)(Bp + (((t * KC) + kc) * 32 + lane) * 16);
            acc[t] = __builtin_amdgcn_wmma_f32_16x16x32_bf16(
                false, a, false, b, (short)0, acc[t], false, false);
        }
    }
}

// ---------------------------------------------------------------------------
// Weight packing into per-lane B fragments (bf16, zero-padded K).
// ---------------------------------------------------------------------------
__global__ void pack_b_kernel(const float* __restrict__ W, __bf16* __restrict__ Bp,
                              int Ktrue, int F, int ntiles, int kchunks) {
    int total = ntiles * kchunks * 32 * 16;
    for (int idx = blockIdx.x * blockDim.x + threadIdx.x; idx < total;
         idx += gridDim.x * blockDim.x) {
        int e    = idx & 15;
        int lane = (idx >> 4) & 31;
        int kc   = (idx >> 9) % kchunks;
        int t    = idx / (512 * kchunks);
        int n    = lane & 15;
        int kh   = lane >> 4;
        int k    = kc * 32 + 16 * kh + e;
        int f    = t * 16 + n;
        float v  = (k < Ktrue) ? W[(long)k * F + f] : 0.f;
        Bp[idx]  = (__bf16)v;
    }
}

__global__ void zero_f32_kernel(float* p, long n) {
    long idx = (long)blockIdx.x * blockDim.x + threadIdx.x;
    if (idx < n) p[idx] = 0.f;
}

// ---------------------------------------------------------------------------
// Embedding: atom = atom_fea @ W_emb + b_emb   (100000 x 92 @ 92 x 64)
// Writes fp32 atom buffer + bf16 mirror for the conv gathers.
// ---------------------------------------------------------------------------
__global__ __launch_bounds__(256) void emb_gemm_kernel(
    const float* __restrict__ atom_fea, const __bf16* __restrict__ Bp,
    const float* __restrict__ b_emb, float* __restrict__ atomF,
    __bf16* __restrict__ atomB) {
    __shared__ __bf16 ldsA[ROWS_BLK * EMB_LDS_STRIDE];
    const int tid = threadIdx.x;
    const long rBase = (long)blockIdx.x * ROWS_BLK;

    for (int idx = tid; idx < ROWS_BLK * EMB_KP; idx += 256) {
        int r = idx / EMB_KP, k = idx - r * EMB_KP;
        long row = rBase + r;
        float v = (row < N_ATOMS && k < EMB_K) ? atom_fea[row * ORIG_F + k] : 0.f;
        ldsA[r * EMB_LDS_STRIDE + k] = (__bf16)v;
    }
    __syncthreads();

    v8f acc[EMB_NT];
#pragma unroll
    for (int t = 0; t < EMB_NT; ++t)
#pragma unroll
        for (int e = 0; e < 8; ++e) acc[t][e] = 0.f;

    wmma_core<EMB_NT, EMB_KC>(ldsA, EMB_LDS_STRIDE, Bp, acc);

    const int lane = tid & 31, wave = tid >> 5;
    const int c0 = lane & 15, rsub = (lane >> 4) * 8;
#pragma unroll
    for (int t = 0; t < EMB_NT; ++t) {
        int c = t * 16 + c0;
        float bb = b_emb[c];
#pragma unroll
        for (int v = 0; v < 8; ++v) {
            long row = rBase + wave * 16 + rsub + v;
            if (row < N_ATOMS) {
                float y = acc[t][v] + bb;
                atomF[row * ATOM_F + c] = y;
                atomB[row * ATOM_F + c] = (__bf16)y;
            }
        }
    }
}

// ---------------------------------------------------------------------------
// Conv staging: gather [self(64) | nbr(64) | nbr_fea(41) | zeros] into LDS.
// Self/neighbor regions are already bf16 in atomB and 16B-aligned, so they
// move as raw 16-byte v8bf copies (no conversion). Only nbr_fea converts.
// One chunk = 8 bf16 elements; 24 chunks per row, 12 iterations per thread.
// ---------------------------------------------------------------------------
__device__ __forceinline__ void conv_stage(__bf16* ldsA, long pBase,
                                           const __bf16* __restrict__ atomB,
                                           const float* __restrict__ nbr_fea,
                                           const int* __restrict__ nbr_idx) {
    const int NCHUNK = CONV_KP / 8;  // 24
    for (int idx = threadIdx.x; idx < ROWS_BLK * NCHUNK; idx += 256) {
        int r = idx / NCHUNK, c = idx - r * NCHUNK;
        int k0 = c * 8;
        long p = pBase + r;
        int i = (int)(p / M_NBR);
        int m = (int)(p - (long)i * M_NBR);
        v8bf out;
        if (k0 < ATOM_F) {
            out = *(const v8bf*)(atomB + (long)i * ATOM_F + k0);
        } else if (k0 < 2 * ATOM_F) {
            int j = nbr_idx[i * M_NBR + m];
            out = *(const v8bf*)(atomB + (long)j * ATOM_F + (k0 - ATOM_F));
        } else {
            const float* src = nbr_fea + ((long)i * M_NBR + m) * NBR_F;
            int kb = k0 - 2 * ATOM_F;  // 0, 8, .., 56
#pragma unroll
            for (int e = 0; e < 8; ++e) {
                int k = kb + e;
                out[e] = (k < NBR_F) ? (__bf16)src[k] : (__bf16)0.f;
            }
        }
        *(v8bf*)(ldsA + r * CONV_LDS_STRIDE + k0) = out;
    }
    // hint the next block's neighbor features into cache (global_prefetch_b8)
    long pn = pBase + ROWS_BLK + threadIdx.x;
    if (pn < NPAIR) __builtin_prefetch(nbr_fea + pn * NBR_F, 0, 1);
}

// Pass 1: GEMM, accumulate per-column sum / sumsq for bn1 (no store of gated).
__global__ __launch_bounds__(256) void conv_gemm_stats_kernel(
    const __bf16* __restrict__ atomB, const float* __restrict__ nbr_fea,
    const int* __restrict__ nbr_idx, const __bf16* __restrict__ Bp,
    float* __restrict__ gsum, float* __restrict__ gsq) {
    __shared__ __bf16 ldsA[ROWS_BLK * CONV_LDS_STRIDE];
    __shared__ float s_sum[128], s_sq[128];
    const int tid = threadIdx.x;
    if (tid < 128) { s_sum[tid] = 0.f; s_sq[tid] = 0.f; }
    const long pBase = (long)blockIdx.x * ROWS_BLK;
    conv_stage(ldsA, pBase, atomB, nbr_fea, nbr_idx);
    __syncthreads();

    v8f acc[CONV_NT];
#pragma unroll
    for (int t = 0; t < CONV_NT; ++t)
#pragma unroll
        for (int e = 0; e < 8; ++e) acc[t][e] = 0.f;

    wmma_core<CONV_NT, CONV_KC>(ldsA, CONV_LDS_STRIDE, Bp, acc);

    const int c0 = tid & 15;
#pragma unroll
    for (int t = 0; t < CONV_NT; ++t) {
        float ls = 0.f, lq = 0.f;
#pragma unroll
        for (int v = 0; v < 8; ++v) { float x = acc[t][v]; ls += x; lq += x * x; }
        atomicAdd(&s_sum[t * 16 + c0], ls);
        atomicAdd(&s_sq[t * 16 + c0], lq);
    }
    __syncthreads();
    if (tid < 128) {
        atomicAdd(&gsum[tid], s_sum[tid]);
        atomicAdd(&gsq[tid], s_sq[tid]);
    }
}

// Pass 2: GEMM again, fused bn1 affine + W_full bias + sigmoid*softplus gate,
// scatter-sum over the 12 neighbors into summed[atom][64].
__global__ __launch_bounds__(256) void conv_gemm_apply_kernel(
    const __bf16* __restrict__ atomB, const float* __restrict__ nbr_fea,
    const int* __restrict__ nbr_idx, const __bf16* __restrict__ Bp,
    const float* __restrict__ sc1, const float* __restrict__ sh1,
    float* __restrict__ summed) {
    __shared__ __bf16 ldsA[ROWS_BLK * CONV_LDS_STRIDE];
    const int tid = threadIdx.x;
    const long pBase = (long)blockIdx.x * ROWS_BLK;
    conv_stage(ldsA, pBase, atomB, nbr_fea, nbr_idx);
    __syncthreads();

    v8f acc[CONV_NT];
#pragma unroll
    for (int t = 0; t < CONV_NT; ++t)
#pragma unroll
        for (int e = 0; e < 8; ++e) acc[t][e] = 0.f;

    wmma_core<CONV_NT, CONV_KC>(ldsA, CONV_LDS_STRIDE, Bp, acc);

    const int lane = tid & 31, wave = tid >> 5;
    const int c0 = lane & 15, rsub = (lane >> 4) * 8;
#pragma unroll
    for (int t = 0; t < 4; ++t) {
        int cf = t * 16 + c0;                 // filter column (0..63), core = cf+64
        float sf = sc1[cf],      hf = sh1[cf];
        float sg = sc1[cf + 64], hg = sh1[cf + 64];
#pragma unroll
        for (int v = 0; v < 8; ++v) {
            float f = acc[t][v] * sf + hf;
            float g = acc[t + 4][v] * sg + hg;
            float val = sigmoidf(f) * softplusf(g);
            long p = pBase + wave * 16 + rsub + v;
            int i = (int)(p / M_NBR);
            atomicAdd(&summed[(long)i * ATOM_F + cf], val);
        }
    }
}

// Per-feature affine finalize for batchnorm: sc = g*rsqrt(var+eps).
// Stats are of the *pre-bias* GEMM output; the linear bias shifts the mean
// only (variance unchanged), so it folds into the affine here.
__global__ void bn_finalize_kernel(const float* gsum, const float* gsq,
                                   const float* bias, const float* gamma,
                                   const float* beta, float cntInv, int n,
                                   float* sc, float* sh) {
    int c = blockIdx.x * blockDim.x + threadIdx.x;
    if (c >= n) return;
    float b   = bias ? bias[c] : 0.f;
    float mu0 = gsum[c] * cntInv;              // mean of pre-bias output
    float var = gsq[c] * cntInv - mu0 * mu0;   // bias shift doesn't change var
    float mu  = mu0 + b;
    float s = gamma[c] * rsqrtf(var + BN_EPS);
    sc[c] = s;
    // applied to pre-bias acc:  (acc + b - mu)*s + beta
    sh[c] = beta[c] + (b - mu) * s;
}

// Column stats over an (rows x 64) fp32 matrix (grid stride is mult of 64).
__global__ __launch_bounds__(256) void colstats64_kernel(
    const float* __restrict__ X, long n, float* gsum, float* gsq) {
    __shared__ float ssum[64], ssq[64];
    const int tid = threadIdx.x;
    if (tid < 64) { ssum[tid] = 0.f; ssq[tid] = 0.f; }
    __syncthreads();
    const long stride = (long)gridDim.x * blockDim.x; // 512*256 -> %64 == 0
    float ls = 0.f, lq = 0.f;
    for (long idx = (long)blockIdx.x * blockDim.x + tid; idx < n; idx += stride) {
        float x = X[idx];
        ls += x; lq += x * x;
    }
    atomicAdd(&ssum[tid & 63], ls);
    atomicAdd(&ssq[tid & 63], lq);
    __syncthreads();
    if (tid < 64) { atomicAdd(&gsum[tid], ssum[tid]); atomicAdd(&gsq[tid], ssq[tid]); }
}

// atom = softplus(atom + bn2(summed)); refresh bf16 mirror.
__global__ void conv_update_kernel(const float* __restrict__ summed,
                                   const float* __restrict__ sc2,
                                   const float* __restrict__ sh2,
                                   float* __restrict__ atomF,
                                   __bf16* __restrict__ atomB, long n) {
    long idx = (long)blockIdx.x * blockDim.x + threadIdx.x;
    if (idx >= n) return;
    int f = (int)(idx & 63);
    float s = summed[idx] * sc2[f] + sh2[f];
    float a = softplusf(atomF[idx] + s);
    atomF[idx] = a;
    atomB[idx] = (__bf16)a;
}

// Segment-sum pooling.
__global__ void pool_accum_kernel(const float* __restrict__ atomF,
                                  const int* __restrict__ cidx,
                                  float* __restrict__ seg, float* __restrict__ cnt,
                                  long n) {
    long idx = (long)blockIdx.x * blockDim.x + threadIdx.x;
    if (idx >= n) return;
    int i = (int)(idx >> 6), f = (int)(idx & 63);
    int c = cidx[i];
    atomicAdd(&seg[(long)c * ATOM_F + f], atomF[idx]);
    if (f == 0) atomicAdd(&cnt[c], 1.f);
}

__global__ void pool_finish_kernel(const float* __restrict__ seg,
                                   const float* __restrict__ cnt,
                                   float* __restrict__ pool, int n) {
    int idx = blockIdx.x * blockDim.x + threadIdx.x;
    if (idx >= n) return;
    int i = idx >> 6;
    pool[idx] = softplusf(seg[idx] / fmaxf(cnt[i], 1.f));
}

// Scalar fp32 head GEMM with fused activation (0=none, 1=relu, 2=softplus).
__global__ void head_gemm_kernel(const float* __restrict__ X,
                                 const float* __restrict__ W,
                                 const float* __restrict__ b,
                                 float* __restrict__ Y,
                                 int M, int K, int N, int act) {
    int idx = blockIdx.x * blockDim.x + threadIdx.x;
    if (idx >= M * N) return;
    int i = idx / N, j = idx - i * N;
    float s = b[j];
    const float* x = X + (long)i * K;
    for (int k = 0; k < K; ++k) s += x[k] * W[(long)k * N + j];
    if (act == 1) s = fmaxf(s, 0.f);
    else if (act == 2) s = softplusf(s);
    Y[idx] = s;
}

__global__ void concat_h0_kernel(const float* __restrict__ crys,
                                 const float* __restrict__ sput,
                                 float* __restrict__ h0, int n) {
    int idx = blockIdx.x * blockDim.x + threadIdx.x;
    if (idx >= n) return;
    int i = idx / (H_F + SPUT_F), j = idx - i * (H_F + SPUT_F);
    h0[idx] = (j < H_F) ? crys[i * H_F + j] : sput[i * SPUT_F + (j - H_F)];
}

// ---------------------------------------------------------------------------
// Host launcher.
// ---------------------------------------------------------------------------
static inline int cdiv(long a, int b) { return (int)((a + b - 1) / b); }

extern "C" void kernel_launch(void* const* d_in, const int* in_sizes, int n_in,
                              void* d_out, int out_size, void* d_ws, size_t ws_size,
                              hipStream_t stream) {
    (void)n_in; (void)out_size; (void)ws_size;
    const float* atom_fea = (const float*)d_in[0];
    const float* nbr_fea  = (const float*)d_in[1];
    const float* sput     = (const float*)d_in[2];

    const float *W_emb, *b_emb, *W_c2f, *b_c2f, *W_a1, *b_a1, *W_a2, *b_a2,
                *W_a3, *b_a3, *W_out, *b_out;
    const float *cW[3], *cb[3], *cg1[3], *cb1[3], *cg2[3], *cb2[3];

    auto F = [&](int i) { return (const float*)d_in[i]; };
    if (in_sizes[3] == (H_F + SPUT_F) * 200) {
        // JAX pytree flatten: dict keys sorted (W_* < b_* < convs)
        W_a1 = F(3); W_a2 = F(4); W_a3 = F(5); W_c2f = F(6); W_emb = F(7); W_out = F(8);
        b_a1 = F(9); b_a2 = F(10); b_a3 = F(11); b_c2f = F(12); b_emb = F(13); b_out = F(14);
        for (int l = 0; l < 3; ++l) {
            int base = 15 + 6 * l; // W_full, b_full, bn1_b, bn1_g, bn2_b, bn2_g
            cW[l] = F(base); cb[l] = F(base + 1);
            cb1[l] = F(base + 2); cg1[l] = F(base + 3);
            cb2[l] = F(base + 4); cg2[l] = F(base + 5);
        }
    } else {
        // insertion order
        W_emb = F(3); b_emb = F(4);
        for (int l = 0; l < 3; ++l) {
            int base = 5 + 6 * l; // W_full, b_full, bn1_g, bn1_b, bn2_g, bn2_b
            cW[l] = F(base); cb[l] = F(base + 1);
            cg1[l] = F(base + 2); cb1[l] = F(base + 3);
            cg2[l] = F(base + 4); cb2[l] = F(base + 5);
        }
        W_c2f = F(23); b_c2f = F(24);
        W_a1 = F(25); b_a1 = F(26); W_a2 = F(27); b_a2 = F(28);
        W_a3 = F(29); b_a3 = F(30); W_out = F(31); b_out = F(32);
    }
    const int* nbr_idx  = (const int*)d_in[33];
    const int* crys_idx = (const int*)d_in[34];

    // Workspace carve-up (~78 MB total)
    char* ws = (char*)d_ws;
    size_t cur = 0;
    auto alloc = [&](size_t bytes) { size_t o = cur; cur = (cur + bytes + 255) & ~(size_t)255; return o; };
    float*  atomF  = (float*) (ws + alloc((size_t)N_ATOMS * ATOM_F * 4));
    __bf16* atomB  = (__bf16*)(ws + alloc((size_t)N_ATOMS * ATOM_F * 2));
    float*  summed = (float*) (ws + alloc((size_t)N_ATOMS * ATOM_F * 4));
    float*  stats  = (float*) (ws + alloc(768 * 4));
    float *gsum1 = stats, *gsq1 = stats + 128, *sc1 = stats + 256, *sh1 = stats + 384;
    float *gsum2 = stats + 512, *gsq2 = stats + 576, *sc2 = stats + 640, *sh2 = stats + 704;
    __bf16* BpEmb = (__bf16*)(ws + alloc((size_t)EMB_NT * EMB_KC * 32 * 16 * 2));
    __bf16* BpConv[3];
    for (int l = 0; l < 3; ++l)
        BpConv[l] = (__bf16*)(ws + alloc((size_t)CONV_NT * CONV_KC * 32 * 16 * 2));
    float* seg  = (float*)(ws + alloc((size_t)N_CRYS * ATOM_F * 4));
    float* cnt  = (float*)(ws + alloc((size_t)N_CRYS * 4));
    float* pool = (float*)(ws + alloc((size_t)N_CRYS * ATOM_F * 4));
    float* crys = (float*)(ws + alloc((size_t)N_CRYS * H_F * 4));
    float* h0   = (float*)(ws + alloc((size_t)N_CRYS * (H_F + SPUT_F) * 4));
    float* h1   = (float*)(ws + alloc((size_t)N_CRYS * 200 * 4));
    float* h2   = (float*)(ws + alloc((size_t)N_CRYS * 400 * 4));
    float* h3   = (float*)(ws + alloc((size_t)N_CRYS * 800 * 4));

    // Pack weights into WMMA B-fragment layouts
    pack_b_kernel<<<cdiv(EMB_NT * EMB_KC * 512, 256), 256, 0, stream>>>(
        W_emb, BpEmb, EMB_K, ATOM_F, EMB_NT, EMB_KC);
    for (int l = 0; l < 3; ++l)
        pack_b_kernel<<<cdiv(CONV_NT * CONV_KC * 512, 256), 256, 0, stream>>>(
            cW[l], BpConv[l], CONV_K, 2 * ATOM_F, CONV_NT, CONV_KC);

    // Embedding
    emb_gemm_kernel<<<EMB_BLOCKS, 256, 0, stream>>>(atom_fea, BpEmb, b_emb, atomF, atomB);

    const long nAF = (long)N_ATOMS * ATOM_F;
    for (int l = 0; l < 3; ++l) {
        zero_f32_kernel<<<1, 256, 0, stream>>>(gsum1, 256);          // gsum1+gsq1
        zero_f32_kernel<<<1, 128, 0, stream>>>(gsum2, 128);          // gsum2+gsq2
        zero_f32_kernel<<<cdiv(nAF, 256), 256, 0, stream>>>(summed, nAF);

        conv_gemm_stats_kernel<<<CONV_BLOCKS, 256, 0, stream>>>(
            atomB, nbr_fea, nbr_idx, BpConv[l], gsum1, gsq1);
        bn_finalize_kernel<<<1, 128, 0, stream>>>(
            gsum1, gsq1, cb[l], cg1[l], cb1[l], 1.f / (float)NPAIR, 128, sc1, sh1);
        conv_gemm_apply_kernel<<<CONV_BLOCKS, 256, 0, stream>>>(
            atomB, nbr_fea, nbr_idx, BpConv[l], sc1, sh1, summed);
        colstats64_kernel<<<512, 256, 0, stream>>>(summed, nAF, gsum2, gsq2);
        bn_finalize_kernel<<<1, 64, 0, stream>>>(
            gsum2, gsq2, nullptr, cg2[l], cb2[l], 1.f / (float)N_ATOMS, 64, sc2, sh2);
        conv_update_kernel<<<cdiv(nAF, 256), 256, 0, stream>>>(
            summed, sc2, sh2, atomF, atomB, nAF);
    }

    // Pooling + head
    zero_f32_kernel<<<cdiv((long)N_CRYS * ATOM_F, 256), 256, 0, stream>>>(seg, (long)N_CRYS * ATOM_F);
    zero_f32_kernel<<<cdiv(N_CRYS, 256), 256, 0, stream>>>(cnt, N_CRYS);
    pool_accum_kernel<<<cdiv(nAF, 256), 256, 0, stream>>>(atomF, crys_idx, seg, cnt, nAF);
    pool_finish_kernel<<<cdiv(N_CRYS * ATOM_F, 256), 256, 0, stream>>>(seg, cnt, pool, N_CRYS * ATOM_F);

    head_gemm_kernel<<<cdiv(N_CRYS * H_F, 256), 256, 0, stream>>>(
        pool, W_c2f, b_c2f, crys, N_CRYS, ATOM_F, H_F, 2);
    concat_h0_kernel<<<cdiv(N_CRYS * (H_F + SPUT_F), 256), 256, 0, stream>>>(
        crys, sput, h0, N_CRYS * (H_F + SPUT_F));
    head_gemm_kernel<<<cdiv(N_CRYS * 200, 256), 256, 0, stream>>>(
        h0, W_a1, b_a1, h1, N_CRYS, H_F + SPUT_F, 200, 1);
    head_gemm_kernel<<<cdiv(N_CRYS * 400, 256), 256, 0, stream>>>(
        h1, W_a2, b_a2, h2, N_CRYS, 200, 400, 1);
    head_gemm_kernel<<<cdiv(N_CRYS * 800, 256), 256, 0, stream>>>(
        h2, W_a3, b_a3, h3, N_CRYS, 400, 800, 1);
    head_gemm_kernel<<<cdiv(N_CRYS, 256), 256, 0, stream>>>(
        h3, W_out, b_out, (float*)d_out, N_CRYS, 800, 1, 0);
}
